// MambaLayer_88545045774369
// MI455X (gfx1250) — matrix operand order repaired and checked by reading, not compiled, not run on hardware
//
#include <hip/hip_runtime.h>
#include <hip/hip_bf16.h>

// ---------- CDNA5 WMMA types ----------
typedef __attribute__((ext_vector_type(16))) _Float16 v16h;
typedef __attribute__((ext_vector_type(8)))  _Float16 v8h;
typedef __attribute__((ext_vector_type(8)))  float    v8f;
typedef __attribute__((ext_vector_type(4)))  unsigned int u32x4;
typedef __attribute__((ext_vector_type(8)))  int i32x8;
typedef __attribute__((ext_vector_type(4)))  int i32x4;

#define HAVE_TDM  __has_builtin(__builtin_amdgcn_tensor_load_to_lds)
#define HAVE_TCNT __has_builtin(__builtin_amdgcn_s_wait_tensorcnt)

// Problem constants (from reference)
#define BB      2
#define CIN     16
#define CDIM    64
#define DINNER  128
#define LSEQ    8192          // 32*32*8
#define BL      16384         // BB*LSEQ
#define K3      5488          // 16*343
#define K3P     5504          // padded to 64
#define NK64    (K3P / 64)    // 86 outer K chunks
#define NSTATE  16
#define DTRANK  4
#define DBLW    48            // padded 36 -> 48

__device__ __forceinline__ v8f wmma16(v16h a, v16h b, v8f c) {
    return __builtin_amdgcn_wmma_f32_16x16x32_f16(false, a, false, b, (short)0, c, false, false);
}

// A fragment: 16x32 f16, row-major source, ld in halves (ISA 7.12.2 layout).
__device__ __forceinline__ v16h ldA(const _Float16* base, int ld, int mbase, int k0, int lane) {
    int row = mbase + (lane & 15);
    const _Float16* p = base + (size_t)row * ld + k0 + ((lane >> 4) << 3);
    v8h lo = *(const v8h*)p;
    v8h hi = *(const v8h*)(p + 16);
    return __builtin_shufflevector(lo, hi, 0,1,2,3,4,5,6,7,8,9,10,11,12,13,14,15);
}

// B fragment: 32x16 f16 from (N,K) row-major, two 16B loads per lane (same K layout as A).
__device__ __forceinline__ v16h ldB(const _Float16* base, int ld, int nbase, int k0, int lane) {
    int row = nbase + (lane & 15);
    const _Float16* p = base + (size_t)row * ld + k0 + ((lane >> 4) << 3);
    v8h lo = *(const v8h*)p;
    v8h hi = *(const v8h*)(p + 16);
    return __builtin_shufflevector(lo, hi, 0,1,2,3,4,5,6,7,8,9,10,11,12,13,14,15);
}

__device__ __forceinline__ float silu(float x) { return x / (1.f + expf(-x)); }

// ---------- weight convert / pad to f16 (N,K) ----------
__global__ void k_pad_f16(const float* __restrict__ src, _Float16* __restrict__ dst,
                          int rows, int scols, int dcols, int drows) {
    int i = blockIdx.x * blockDim.x + threadIdx.x;
    int total = drows * dcols;
    if (i >= total) return;
    int r = i / dcols, c = i - r * dcols;
    float v = (r < rows && c < scols) ? src[(size_t)r * scols + c] : 0.f;
    dst[i] = (_Float16)v;
}

// ---------- im2col K-table: k_part offset + packed tap coords ----------
__global__ void k_ktable(int* __restrict__ tbl) {   // int2 per K: {kpart, kd|kh<<8|kw<<16|valid<<24}
    int K = blockIdx.x * blockDim.x + threadIdx.x;
    if (K >= K3P) return;
    int kpart = 0, packed = 0;
    if (K < K3) {
        int ic = K / 343; int r = K - ic * 343;
        int kd = r / 49;  r -= kd * 49;
        int kh = r / 7;   int kw = r - kh * 7;
        kpart  = ic * 65536 + kd * 1024 + kh * 16 + kw;
        packed = kd | (kh << 8) | (kw << 16) | (1 << 24);
    }
    tbl[2 * K]     = kpart;
    tbl[2 * K + 1] = packed;
}

// ---------- conv3d: implicit GEMM, WMMA + double-buffered TDM weight staging ----------
// grid.x = BL/32 ; block = 256 (8 waves: 2 M tiles x 4 N tiles). K step = 64.
__global__ void k_conv3d(const float* __restrict__ x, const _Float16* __restrict__ w,
                         const float* __restrict__ bias, const int* __restrict__ tbl,
                         float* __restrict__ h) {
    __shared__ __align__(32) _Float16 At[2][32 * 64];   // A tiles: 32 rows x 64 K
    __shared__ __align__(32) _Float16 Bt[2][64 * 64];   // weight tiles: 64 N x 64 K (TDM dest)
    const int tid   = threadIdx.x;
    const int lane  = tid & 31;
    const int wv    = tid >> 5;              // 0..7
    const int ntile = (wv & 3) * 16;
    const int mtile = (wv >> 2) * 16;
    const int mbase = blockIdx.x * 32;

    // per-thread gather coords (loop invariant): row ml, K sub-range kkb..kkb+7
    const int ml  = tid >> 3;                // 0..31
    const int kkb = (tid & 7) * 8;           // 0..56
    const int M   = mbase + ml;
    const int b   = M >> 13, l = M & (LSEQ - 1);
    const int od2 = 2 * (l >> 8) - 3, oh2 = 2 * ((l >> 3) & 31) - 3, ow2 = 2 * (l & 7) - 3;
    const int m_off = b * CIN * 65536 + od2 * 1024 + oh2 * 16 + ow2;

#if HAVE_TDM
    const unsigned ldsB0 = (unsigned)(uintptr_t)(void*)&Bt[0][0];   // LDS byte offset
    const unsigned ldsB1 = (unsigned)(uintptr_t)(void*)&Bt[1][0];
    const unsigned long long gbase = (unsigned long long)(uintptr_t)w;
#endif

    // issue weight-tile fetch for K chunk k0 into buffer buf
    auto fetchB = [&](int k0, int buf) {
#if HAVE_TDM
        if (wv == 0) {
            // D# per CDNA5 ISA
            unsigned long long ga = gbase + (unsigned long long)k0 * 2ull;
            u32x4 g0 = { 1u, buf ? ldsB1 : ldsB0,
                         (unsigned)(ga & 0xFFFFFFFFull),
                         (unsigned)((ga >> 32) & 0x01FFFFFFull) | 0x80000000u };   // type=2
            i32x8 g1 = { 0x00010000,            // data_size=2B
                         (int)(K3P << 16),      // tensor_dim0[15:0]
                         (int)(64 << 16),       // tensor_dim1[15:0]
                         (int)(64 << 16),       // tile_dim0
                         64,                    // tile_dim1
                         K3P,                   // tensor_dim0_stride[31:0]
                         0, 0 };
            i32x4 gz = { 0, 0, 0, 0 };
#if defined(__clang_major__) && __clang_major__ >= 23
            i32x8 gz8 = { 0, 0, 0, 0, 0, 0, 0, 0 };
            __builtin_amdgcn_tensor_load_to_lds(g0, g1, gz, gz, gz8, 0);
#else
            __builtin_amdgcn_tensor_load_to_lds(g0, g1, gz, gz, 0);
#endif
        }
#else
        for (int t = tid; t < 64 * 64; t += 256)
            Bt[buf][t] = w[(size_t)(t >> 6) * K3P + k0 + (t & 63)];
#endif
    };

    // branchless im2col gather of this thread's 8 A-tile elements for chunk k0
    auto gatherA = [&](int k0, int buf) {
        const int4* t4 = (const int4*)(tbl + 2 * (k0 + kkb));   // 64B aligned
        int4 e0 = t4[0], e1 = t4[1], e2 = t4[2], e3 = t4[3];
        int kp[8] = { e0.x, e0.z, e1.x, e1.z, e2.x, e2.z, e3.x, e3.z };
        int pk[8] = { e0.y, e0.w, e1.y, e1.w, e2.y, e2.w, e3.y, e3.w };
        v8h pack;
#pragma unroll
        for (int i = 0; i < 8; ++i) {
            int id = od2 + (pk[i] & 0xFF);
            int ih = oh2 + ((pk[i] >> 8) & 0xFF);
            int iw = ow2 + ((pk[i] >> 16) & 0xFF);
            bool ok = (pk[i] & (1 << 24)) && (unsigned)id < 64u &&
                      (unsigned)ih < 64u && (unsigned)iw < 16u;
            float v = x[ok ? (m_off + kp[i]) : 0];   // unconditional load, clamped addr
            pack[i] = (_Float16)(ok ? v : 0.f);
        }
        *(v8h*)(&At[buf][ml * 64 + kkb]) = pack;     // single ds_store_b128
    };

    v8f acc = {};
    // prologue: fill buffer 0
    fetchB(0, 0);
    gatherA(0, 0);
#if HAVE_TDM && HAVE_TCNT
    if (wv == 0) __builtin_amdgcn_s_wait_tensorcnt((short)0);
#endif
    __syncthreads();

    for (int kc = 0; kc < NK64; ++kc) {
        const int cur = kc & 1, nxt = cur ^ 1;
        // issue current-chunk LDS reads first ...
        v16h a0 = ldA(At[cur], 64, mtile, 0,  lane);
        v16h a1 = ldA(At[cur], 64, mtile, 32, lane);
        v16h b0 = ldB(Bt[cur], 64, ntile, 0,  lane);
        v16h b1 = ldB(Bt[cur], 64, ntile, 32, lane);
        // ... then start next-chunk staging so its latency hides under the WMMAs
        if (kc + 1 < NK64) {
            fetchB((kc + 1) * 64, nxt);
            gatherA((kc + 1) * 64, nxt);
        }
        acc = wmma16(a0, b0, acc);
        acc = wmma16(a1, b1, acc);
#if HAVE_TDM && HAVE_TCNT
        if (wv == 0) __builtin_amdgcn_s_wait_tensorcnt((short)0);
#endif
        __syncthreads();   // publish nxt buffers / retire cur reads
    }

    int n = ntile + (lane & 15);
    float bv = bias[n];
#pragma unroll
    for (int r = 0; r < 8; ++r) {
        int Mo = mbase + mtile + r + ((lane >> 4) << 3);
        int bo = Mo >> 13, lo = Mo & (LSEQ - 1);
        h[((size_t)(bo * CDIM + n)) * LSEQ + lo] = acc[r] + bv;
    }
}

// ---------- LayerNorm over C=64, wave per row, output f16 (M,K) ----------
__global__ void k_ln(const float* __restrict__ h, const float* __restrict__ g,
                     const float* __restrict__ bta, _Float16* __restrict__ xn) {
    int wid  = (blockIdx.x * blockDim.x + threadIdx.x) >> 5;
    int lane = threadIdx.x & 31;
    if (wid >= BL) return;
    int b = wid >> 13, l = wid & (LSEQ - 1);
    const float* hp = h + (size_t)b * CDIM * LSEQ + l;
    float v0 = hp[(size_t)lane * LSEQ];
    float v1 = hp[(size_t)(lane + 32) * LSEQ];
    float s = v0 + v1, s2 = v0 * v0 + v1 * v1;
#pragma unroll
    for (int o = 16; o; o >>= 1) { s += __shfl_xor(s, o, 32); s2 += __shfl_xor(s2, o, 32); }
    float mu  = s * (1.f / 64.f);
    float var = s2 * (1.f / 64.f) - mu * mu;
    float rs  = rsqrtf(var + 1e-5f);
    _Float16* xp = xn + (size_t)wid * CDIM;
    xp[lane]      = (_Float16)((v0 - mu) * rs * g[lane] + bta[lane]);
    xp[lane + 32] = (_Float16)((v1 - mu) * rs * g[lane + 32] + bta[lane + 32]);
}

// ---------- in_proj GEMM: (BL,64) x (256,64)^T, split xi/z ----------
__global__ void k_inproj(const _Float16* __restrict__ A, const _Float16* __restrict__ Bw,
                         float* __restrict__ xi, float* __restrict__ z) {
    const int lane = threadIdx.x & 31;
    const int wv   = threadIdx.x >> 5;   // 0..7
    const int mbase = blockIdx.x * 16;
    const int n0 = wv * 32, n1 = wv * 32 + 16;
    v8f acc0 = {}, acc1 = {};
#pragma unroll
    for (int kc = 0; kc < 2; ++kc) {
        int k0 = kc * 32;
        v16h a  = ldA(A, CDIM, mbase, k0, lane);
        v16h b0 = ldB(Bw, CDIM, n0, k0, lane);
        v16h b1 = ldB(Bw, CDIM, n1, k0, lane);
        acc0 = wmma16(a, b0, acc0);
        acc1 = wmma16(a, b1, acc1);
    }
    int nc0 = n0 + (lane & 15), nc1 = n1 + (lane & 15);
#pragma unroll
    for (int r = 0; r < 8; ++r) {
        int M = mbase + r + ((lane >> 4) << 3);
        if (nc0 < DINNER) xi[(size_t)M * DINNER + nc0] = acc0[r];
        else              z [(size_t)M * DINNER + nc0 - DINNER] = acc0[r];
        if (nc1 < DINNER) xi[(size_t)M * DINNER + nc1] = acc1[r];
        else              z [(size_t)M * DINNER + nc1 - DINNER] = acc1[r];
    }
}

// ---------- causal depthwise conv1d + SiLU (sequence order, per dir) ----------
__global__ void k_dwconv(const float* __restrict__ xi, const float* __restrict__ w,
                         const float* __restrict__ bias, _Float16* __restrict__ xc, int dir) {
    int idx = blockIdx.x * blockDim.x + threadIdx.x;
    if (idx >= BL * DINNER) return;
    int d  = idx & (DINNER - 1);
    int bl = idx >> 7;
    int b = bl >> 13, s = bl & (LSEQ - 1);
    float acc = bias[d];
#pragma unroll
    for (int j = 0; j < 4; ++j) {
        int t = s - 3 + j;
        if (t >= 0) {
            int lo = dir ? (LSEQ - 1 - t) : t;
            acc += w[d * 4 + j] * xi[((size_t)(b * LSEQ + lo)) * DINNER + d];
        }
    }
    xc[(size_t)bl * DINNER + d] = (_Float16)silu(acc);
}

// ---------- x_proj GEMM: (BL,128) x (48,128)^T per dir ----------
__global__ void k_xproj(const _Float16* __restrict__ A, const _Float16* __restrict__ Bw,
                        float* __restrict__ dbl) {
    const int lane = threadIdx.x & 31;
    const int wv   = threadIdx.x >> 5;   // 0..2
    const _Float16* Ad = A  + (size_t)blockIdx.y * BL * DINNER;
    const _Float16* Bd = Bw + (size_t)blockIdx.y * DBLW * DINNER;
    float* Dd = dbl + (size_t)blockIdx.y * BL * DBLW;
    const int mbase = blockIdx.x * 16;
    v8f acc = {};
#pragma unroll
    for (int kc = 0; kc < 4; ++kc) {
        v16h a = ldA(Ad, DINNER, mbase, kc * 32, lane);
        v16h b = ldB(Bd, DINNER, wv * 16, kc * 32, lane);
        acc = wmma16(a, b, acc);
    }
    int n = wv * 16 + (lane & 15);
#pragma unroll
    for (int r = 0; r < 8; ++r) {
        int M = mbase + r + ((lane >> 4) << 3);
        Dd[(size_t)M * DBLW + n] = acc[r];
    }
}

// ---------- selective scan: wave per 2 channels; lanes 0..15 = states ----------
__global__ void k_scan(const _Float16* __restrict__ xc, const float* __restrict__ dbl,
                       const float* __restrict__ dtw, const float* __restrict__ dtb,
                       const float* __restrict__ Alog, const float* __restrict__ Dp,
                       float* __restrict__ yacc, int dir) {
    int bid = blockIdx.x;
    int b = bid >> 6, dp = bid & 63;
    int lane = threadIdx.x;
    int n = lane & 15;
    int d = dp * 2 + (lane >> 4);
    const _Float16* xcd = xc  + (size_t)dir * BL * DINNER;
    const float*   dbld = dbl + (size_t)dir * BL * DBLW;
    float An = -expf(Alog[d * NSTATE + n]);
    float w0 = dtw[d * 4 + 0], w1 = dtw[d * 4 + 1], w2 = dtw[d * 4 + 2], w3 = dtw[d * 4 + 3];
    float tb = dtb[d], Dd = Dp[d];
    float hst = 0.f;
    for (int s = 0; s < LSEQ; ++s) {
        size_t bl = (size_t)(b * LSEQ + s);
        float u = (float)xcd[bl * DINNER + d];
        const float* row = dbld + bl * DBLW;
        float dtr = w0 * row[0] + w1 * row[1] + w2 * row[2] + w3 * row[3] + tb;
        float delta = (dtr > 20.f) ? dtr : log1pf(expf(dtr));
        float Bn = row[DTRANK + n], Cn = row[DTRANK + NSTATE + n];
        hst = expf(delta * An) * hst + (delta * u) * Bn;
        float p = hst * Cn;
        p += __shfl_xor(p, 1, 32);
        p += __shfl_xor(p, 2, 32);
        p += __shfl_xor(p, 4, 32);
        p += __shfl_xor(p, 8, 32);
        if (n == 0) {
            int lo = dir ? (LSEQ - 1 - s) : s;
            size_t a = ((size_t)(b * LSEQ + lo)) * DINNER + d;
            float y = p + u * Dd;
            if (dir) yacc[a] += y; else yacc[a] = y;
        }
    }
}

// ---------- gate: yg = (y_f + y_b) * silu(z), emit f16 A-matrix ----------
__global__ void k_gate(const float* __restrict__ yacc, const float* __restrict__ z,
                       _Float16* __restrict__ yg) {
    int i = blockIdx.x * blockDim.x + threadIdx.x;
    if (i >= BL * DINNER) return;
    yg[i] = (_Float16)(yacc[i] * silu(z[i]));
}

// ---------- out_proj GEMM + residual ----------
__global__ void k_outproj(const _Float16* __restrict__ A, const _Float16* __restrict__ Bw,
                          const float* __restrict__ h, float* __restrict__ out) {
    const int lane = threadIdx.x & 31;
    const int wv   = threadIdx.x >> 5;
    const int mbase = blockIdx.x * 16;
    v8f acc = {};
#pragma unroll
    for (int kc = 0; kc < 4; ++kc) {
        v16h a = ldA(A, DINNER, mbase, kc * 32, lane);
        v16h b = ldB(Bw, DINNER, wv * 16, kc * 32, lane);
        acc = wmma16(a, b, acc);
    }
    int c = wv * 16 + (lane & 15);
#pragma unroll
    for (int r = 0; r < 8; ++r) {
        int M = mbase + r + ((lane >> 4) << 3);
        int b = M >> 13, l = M & (LSEQ - 1);
        size_t a2 = ((size_t)(b * CDIM + c)) * LSEQ + l;
        out[a2] = acc[r] + h[a2];
    }
}

extern "C" void kernel_launch(void* const* d_in, const int* in_sizes, int n_in,
                              void* d_out, int out_size, void* d_ws, size_t ws_size,
                              hipStream_t stream) {
    const float* x        = (const float*)d_in[0];
    const float* conv3d_w = (const float*)d_in[1];
    const float* conv3d_b = (const float*)d_in[2];
    const float* ln_g     = (const float*)d_in[3];
    const float* ln_b     = (const float*)d_in[4];
    const float* inproj_w = (const float*)d_in[5];
    const float* outproj_w= (const float*)d_in[6];
    const float* c1w[2]  = { (const float*)d_in[7],  (const float*)d_in[14] };
    const float* c1b[2]  = { (const float*)d_in[8],  (const float*)d_in[15] };
    const float* xpw[2]  = { (const float*)d_in[9],  (const float*)d_in[16] };
    const float* dtw[2]  = { (const float*)d_in[10], (const float*)d_in[17] };
    const float* dtb[2]  = { (const float*)d_in[11], (const float*)d_in[18] };
    const float* alog[2] = { (const float*)d_in[12], (const float*)d_in[19] };
    const float* Dp[2]   = { (const float*)d_in[13], (const float*)d_in[20] };
    float* out = (float*)d_out;

    // ---- workspace carve ----
    char* p = (char*)d_ws;
    char* pend = p + ws_size;
    auto carve = [&](size_t bytes) -> char* {
        char* r = p; p += (bytes + 255) & ~(size_t)255; return r;
    };
    _Float16* w3d  = (_Float16*)carve((size_t)CDIM * K3P * 2);
    _Float16* win  = (_Float16*)carve((size_t)(2 * DINNER) * CDIM * 2);
    _Float16* wxp  = (_Float16*)carve((size_t)2 * DBLW * DINNER * 2);
    _Float16* wout = (_Float16*)carve((size_t)CDIM * DINNER * 2);
    int*      ktbl = (int*)     carve((size_t)K3P * 2 * 4);
    float*    hbuf = (float*)   carve((size_t)BB * CDIM * LSEQ * 4);
    _Float16* xn   = (_Float16*)carve((size_t)BL * CDIM * 2);
    float*    xi   = (float*)   carve((size_t)BL * DINNER * 4);
    float*    zz   = (float*)   carve((size_t)BL * DINNER * 4);
    _Float16* xc   = (_Float16*)carve((size_t)2 * BL * DINNER * 2);
    float*    dbl  = (float*)   carve((size_t)2 * BL * DBLW * 4);
    float*    yacc = (float*)   carve((size_t)BL * DINNER * 4);
    _Float16* yg   = (_Float16*)carve((size_t)BL * DINNER * 2);
    if (p > pend) return;

    // ---- weight conversion + tables ----
    {
        int t0 = CDIM * K3P;
        k_pad_f16<<<(t0 + 255) / 256, 256, 0, stream>>>(conv3d_w, w3d, CDIM, K3, K3P, CDIM);
        int t1 = 2 * DINNER * CDIM;
        k_pad_f16<<<(t1 + 255) / 256, 256, 0, stream>>>(inproj_w, win, 2 * DINNER, CDIM, CDIM, 2 * DINNER);
        int t2 = DBLW * DINNER;
        k_pad_f16<<<(t2 + 255) / 256, 256, 0, stream>>>(xpw[0], wxp, 36, DINNER, DINNER, DBLW);
        k_pad_f16<<<(t2 + 255) / 256, 256, 0, stream>>>(xpw[1], wxp + (size_t)DBLW * DINNER, 36, DINNER, DINNER, DBLW);
        int t3 = CDIM * DINNER;
        k_pad_f16<<<(t3 + 255) / 256, 256, 0, stream>>>(outproj_w, wout, CDIM, DINNER, DINNER, CDIM);
        k_ktable<<<(K3P + 255) / 256, 256, 0, stream>>>(ktbl);
    }

    // ---- pipeline ----
    k_conv3d<<<BL / 32, 256, 0, stream>>>(x, w3d, conv3d_b, ktbl, hbuf);
    k_ln<<<BL / 8, 256, 0, stream>>>(hbuf, ln_g, ln_b, xn);
    k_inproj<<<BL / 16, 256, 0, stream>>>(xn, win, xi, zz);
    for (int dir = 0; dir < 2; ++dir)
        k_dwconv<<<(BL * DINNER + 255) / 256, 256, 0, stream>>>(
            xi, c1w[dir], c1b[dir], xc + (size_t)dir * BL * DINNER, dir);
    {
        dim3 g(BL / 16, 2);
        k_xproj<<<g, 96, 0, stream>>>(xc, wxp, dbl);
    }
    for (int dir = 0; dir < 2; ++dir)
        k_scan<<<BB * 64, 32, 0, stream>>>(xc, dbl, dtw[dir], dtb[dir], alog[dir], Dp[dir], yacc, dir);
    k_gate<<<(BL * DINNER + 255) / 256, 256, 0, stream>>>(yacc, zz, yg);
    k_outproj<<<BL / 16, 128, 0, stream>>>(yg, wout, hbuf, out);
}
